// Linear_27453430956526
// MI455X (gfx1250) — 8-bit matrix operand order repaired and checked by reading, compiled, not run on hardware
//
#include <hip/hip_runtime.h>
#include <hip/hip_bf16.h>

typedef __attribute__((ext_vector_type(16))) int   v16i;
typedef __attribute__((ext_vector_type(8)))  float v8f;

#define FP8MAX 448.0f

__device__ __forceinline__ unsigned int f32_to_e4m3(float f) {
  unsigned int u   = __float_as_uint(f);
  unsigned int sgn = (u >> 31) << 7;
  float a = fabsf(f);
  if (!(a == a)) return sgn | 0x7Fu;
  if (a >= FP8MAX) return sgn | 0x7Eu;
  if (a >= 0.015625f) {
    int e = (int)((u >> 23) & 0xFF) - 127;
    float ulp = __uint_as_float((unsigned int)(e - 3 + 127) << 23);
    float q = rintf(a / ulp) * ulp;
    unsigned int qb = __float_as_uint(q);
    int qe = (int)((qb >> 23) & 0xFF) - 127;
    if (qe > 8) return sgn | 0x7Eu;
    return sgn | (unsigned int)(((qe + 7) << 3) | ((qb >> 20) & 7));
  }
  unsigned int m = (unsigned int)rintf(a * 512.0f);
  return sgn | m;
}

__device__ __forceinline__ unsigned int pack4_e4m3(float a, float b, float c, float d) {
#if __has_builtin(__builtin_amdgcn_cvt_pk_fp8_f32)
  int lo = __builtin_amdgcn_cvt_pk_fp8_f32(a, b, 0, false);
  int pk = __builtin_amdgcn_cvt_pk_fp8_f32(c, d, lo, true);
  return (unsigned int)pk;
#else
  return f32_to_e4m3(a) | (f32_to_e4m3(b) << 8) |
         (f32_to_e4m3(c) << 16) | (f32_to_e4m3(d) << 24);
#endif
}

__device__ __forceinline__ unsigned lds_off(const void* p) {
  return (unsigned)(unsigned long long)p;
}

__device__ __forceinline__ void async_load_b128(const void* gaddr, unsigned ldsoff) {
  asm volatile("global_load_async_to_lds_b128 %0, %1, off"
               :: "v"(ldsoff), "v"(gaddr) : "memory");
}

__device__ __forceinline__ void wait_async0() {
#if __has_builtin(__builtin_amdgcn_s_wait_asynccnt)
  __builtin_amdgcn_s_wait_asynccnt(0);
#else
  asm volatile("s_wait_asynccnt 0x0" ::: "memory");
#endif
}

__global__ __launch_bounds__(256) void act_quant_kernel(
    const float* __restrict__ x, unsigned char* __restrict__ xq,
    float* __restrict__ xs, int M, int K) {
  const int Kb = K >> 7;
  int wave = blockIdx.x * (blockDim.x >> 5) + (threadIdx.x >> 5);
  int lane = threadIdx.x & 31;
  if (wave >= M * Kb) return;
  int m  = wave / Kb;
  int kb = wave - m * Kb;

  const float4 v = *(const float4*)(x + (size_t)m * K + kb * 128 + lane * 4);
  float a = fmaxf(fmaxf(fabsf(v.x), fabsf(v.y)), fmaxf(fabsf(v.z), fabsf(v.w)));
  #pragma unroll
  for (int off = 16; off >= 1; off >>= 1)
    a = fmaxf(a, __shfl_xor(a, off, 32));

  float s = fmaxf(a, 1e-12f) * (1.0f / FP8MAX);
  unsigned int packed = pack4_e4m3(v.x / s, v.y / s, v.z / s, v.w / s);
  *(unsigned int*)(xq + (size_t)m * K + kb * 128 + lane * 4) = packed;
  if (lane == 0) xs[(size_t)m * Kb + kb] = s;
}

__global__ __launch_bounds__(256) void wq_convert_kernel(
    const float* __restrict__ w, unsigned char* __restrict__ wq, long long total4) {
  long long t = (long long)blockIdx.x * blockDim.x + threadIdx.x;
  if (t >= total4) return;
  const float4 v = *(const float4*)(w + t * 4);
  *(unsigned int*)(wq + t * 4) = pack4_e4m3(v.x, v.y, v.z, v.w);
}

#define LSTRIDE 144

__global__ __launch_bounds__(256) void fp8_gemm_kernel(
    const unsigned char* __restrict__ xq, const unsigned char* __restrict__ wq,
    const float* __restrict__ xs, const float* __restrict__ wscale,
    float* __restrict__ out, int M, int N, int K) {
  const int Kb  = K >> 7;
  const int tid = threadIdx.x;
  const int lane = tid & 31;
  const int wid  = tid >> 5;
  const int h    = lane >> 4;
  const int l16  = lane & 15;
  const int m0 = blockIdx.y * 128;
  const int n0 = blockIdx.x * 128;
  const int nb = blockIdx.x;
  const int wrow = (wid & 1) * 64;
  const int wcol = (wid >> 1) * 32;

  __shared__ __align__(16) unsigned char sA[2][128 * LSTRIDE];
  __shared__ __align__(16) unsigned char sB[2][128 * LSTRIDE];
  __shared__ __align__(16) float sXS[32 * 128];

  const int srow = tid >> 1;
  const int scol = (tid & 1) << 6;

  {
    const unsigned char* ga = xq + (size_t)(m0 + srow) * K + scol;
    const unsigned char* gb = wq + (size_t)(n0 + srow) * K + scol;
    unsigned la = lds_off(&sA[0][srow * LSTRIDE + scol]);
    unsigned lb = lds_off(&sB[0][srow * LSTRIDE + scol]);
    #pragma unroll
    for (int i = 0; i < 4; ++i) {
      async_load_b128(ga + i * 16, la + i * 16);
      async_load_b128(gb + i * 16, lb + i * 16);
    }
  }

  for (int i = tid; i < 128 * Kb; i += 256) {
    int mm = i / Kb, kk = i - mm * Kb;
    sXS[kk * 128 + mm] = xs[(size_t)(m0 + mm) * Kb + kk];
  }

  v8f acc[4][2];
  #pragma unroll
  for (int mt = 0; mt < 4; ++mt)
    #pragma unroll
    for (int nt = 0; nt < 2; ++nt)
      acc[mt][nt] = (v8f){};

  wait_async0();
  __syncthreads();

  for (int kb = 0; kb < Kb; ++kb) {
    const int cur = kb & 1;

    if (kb + 1 < Kb) {
      const int nxt = cur ^ 1;
      const unsigned char* ga = xq + (size_t)(m0 + srow) * K + (kb + 1) * 128 + scol;
      const unsigned char* gb = wq + (size_t)(n0 + srow) * K + (kb + 1) * 128 + scol;
      unsigned la = lds_off(&sA[nxt][srow * LSTRIDE + scol]);
      unsigned lb = lds_off(&sB[nxt][srow * LSTRIDE + scol]);
      #pragma unroll
      for (int i = 0; i < 4; ++i) {
        async_load_b128(ga + i * 16, la + i * 16);
        async_load_b128(gb + i * 16, lb + i * 16);
      }
    }

    const float ws = wscale[(size_t)nb * Kb + kb];

    v16i bfrag[2];
    #pragma unroll
    for (int nt = 0; nt < 2; ++nt) {
      const unsigned char* bp = &sB[cur][(wcol + nt * 16 + l16) * LSTRIDE + h * 16];
      #pragma unroll
      for (int g = 0; g < 4; ++g) {
        uint4 q = *(const uint4*)(bp + g * 32);
        bfrag[nt][4 * g + 0] = (int)q.x;
        bfrag[nt][4 * g + 1] = (int)q.y;
        bfrag[nt][4 * g + 2] = (int)q.z;
        bfrag[nt][4 * g + 3] = (int)q.w;
      }
    }

    #pragma unroll
    for (int mt = 0; mt < 4; ++mt) {
      const unsigned char* ap = &sA[cur][(wrow + mt * 16 + l16) * LSTRIDE + h * 16];
      v16i afrag;
      #pragma unroll
      for (int c = 0; c < 8; ++c) {
        uint2 q = *(const uint2*)(ap + (c >> 1) * 32 + (c & 1) * 8);
        afrag[2 * c + 0] = (int)q.x;
        afrag[2 * c + 1] = (int)q.y;
      }
      const float* xp = &sXS[kb * 128 + wrow + mt * 16 + h * 8];
      float4 x0 = *(const float4*)(xp);
      float4 x1 = *(const float4*)(xp + 4);
      float xsws[8] = { x0.x * ws, x0.y * ws, x0.z * ws, x0.w * ws,
                        x1.x * ws, x1.y * ws, x1.z * ws, x1.w * ws };

      #pragma unroll
      for (int nt = 0; nt < 2; ++nt) {
        v8f zero = (v8f){};
        v8f t = __builtin_amdgcn_wmma_f32_16x16x128_fp8_fp8(
            afrag, bfrag[nt], (short)0, zero, false, false);
        #pragma unroll
        for (int r = 0; r < 8; ++r)
          acc[mt][nt][r] = fmaf(xsws[r], t[r], acc[mt][nt][r]);
      }
    }

    wait_async0();
    __syncthreads();
  }

  #pragma unroll
  for (int mt = 0; mt < 4; ++mt)
    #pragma unroll
    for (int nt = 0; nt < 2; ++nt)
      #pragma unroll
      for (int r = 0; r < 8; ++r) {
        int row = m0 + wrow + mt * 16 + h * 8 + r;
        int col = n0 + wcol + nt * 16 + l16;
        out[(size_t)row * N + col] = acc[mt][nt][r];
      }
}

extern "C" void kernel_launch(void* const* d_in, const int* in_sizes, int n_in,
                              void* d_out, int out_size, void* d_ws, size_t ws_size,
                              hipStream_t stream) {
  const float* x  = (const float*)d_in[0];
  const float* w  = (const float*)d_in[1];
  const float* wsc = (const float*)d_in[2];

  const int K = 4096;
  const int M = in_sizes[0] / K;
  const int N = in_sizes[1] / K;
  const int Kb = K / 128;

  unsigned char* xq = (unsigned char*)d_ws;
  unsigned char* wq = xq + (size_t)M * K;
  float* xsbuf = (float*)(wq + (size_t)N * K);

  {
    int waves = M * Kb;
    int blocks = (waves + 7) / 8;
    act_quant_kernel<<<blocks, 256, 0, stream>>>(x, xq, xsbuf, M, K);
  }
  {
    long long total4 = ((long long)N * K) / 4;
    int blocks = (int)((total4 + 255) / 256);
    wq_convert_kernel<<<blocks, 256, 0, stream>>>(w, wq, total4);
  }
  {
    dim3 grid(N / 128, M / 128);
    fp8_gemm_kernel<<<grid, 256, 0, stream>>>(xq, wq, xsbuf, wsc,
                                              (float*)d_out, M, N, K);
  }
}
